// PhaGruMPN3_38405597561075
// MI455X (gfx1250) — matrix operand order repaired and checked by reading, not compiled, not run on hardware
//
#include <hip/hip_runtime.h>
#include <hip/hip_bf16.h>

#define N_ATOMS 250000
#define N_PAIRS 4000000
#define N_MOL   2000
#define MAX_NEI 16
#define MAX_ATOMS_PER_MOL 128
#define HID 32
#define FEAT 8
#define ALL_FDIM 9
#define DEPTH 3

typedef __attribute__((ext_vector_type(16))) _Float16 v16h;
typedef __attribute__((ext_vector_type(8)))  float    v8f;

// ---------------------------------------------------------------------------
// WMMA helper (v_wmma_f32_16x16x32_f16, wave32). Layouts per CDNA5 ISA §7.12.2
// ---------------------------------------------------------------------------
__device__ __forceinline__ v8f wmma_acc(v16h a, v16h b, v8f c) {
    // 8 args: (neg_a, A, neg_b, B, c_mod, C, reuse_a, reuse_b)
    return __builtin_amdgcn_wmma_f32_16x16x32_f16(false, a, false, b, (short)0, c,
                                                  false, false);
}

// Load a 16x32 A tile from a row-major fp32 matrix with row stride 32 (128B).
// A layout: lane L -> row M = L&15; group g = L>>4;
//   element e: K = (e<8 ? 0 : 16) + 8*g + (e&7)
// Per lane that is two contiguous 8-float runs -> 4x b128 loads + cvt.
__device__ __forceinline__ v16h load_A_rowmajor(const float* base, int row0, int lane) {
    const int m = lane & 15;
    const int g = lane >> 4;
    const float4* row = (const float4*)(base + (size_t)(row0 + m) * HID);
    const float4 x0 = row[2 * g];          // K = 8g .. 8g+3
    const float4 x1 = row[2 * g + 1];      // K = 8g+4 .. 8g+7
    const float4 x2 = row[4 + 2 * g];      // K = 16+8g .. 16+8g+3
    const float4 x3 = row[5 + 2 * g];      // K = 16+8g+4 .. 16+8g+7
    v16h a;
    a[0]  = (_Float16)x0.x; a[1]  = (_Float16)x0.y;
    a[2]  = (_Float16)x0.z; a[3]  = (_Float16)x0.w;
    a[4]  = (_Float16)x1.x; a[5]  = (_Float16)x1.y;
    a[6]  = (_Float16)x1.z; a[7]  = (_Float16)x1.w;
    a[8]  = (_Float16)x2.x; a[9]  = (_Float16)x2.y;
    a[10] = (_Float16)x2.z; a[11] = (_Float16)x2.w;
    a[12] = (_Float16)x3.x; a[13] = (_Float16)x3.y;
    a[14] = (_Float16)x3.z; a[15] = (_Float16)x3.w;
    return a;
}

// Pre-packed B tile: 16 contiguous f16 per lane -> one 32B vector load.
__device__ __forceinline__ v16h load_B_packed(const _Float16* P, int tile, int lane) {
    return *(const v16h*)(P + (size_t)tile * 512 + lane * 16);
}

// Store a 16x16 fp32 D tile to a row-major matrix with row stride 32.
// D layout: lane L -> col N = L&15; VGPR r -> row M = r + 8*(L>>4)
__device__ __forceinline__ void store_D(float* base, int row0, int col0, int lane,
                                        v8f d, bool valid) {
    if (!valid) return;
    const int n = lane & 15;
    const int g = lane >> 4;
#pragma unroll
    for (int r = 0; r < 8; ++r) {
        const int m = r + 8 * g;
        base[(size_t)(row0 + m) * HID + col0 + n] = d[r];
    }
}

// ---------------------------------------------------------------------------
// Pack 32x16 B tiles of a row-major f32 weight [ktiles*32 x 32] into the
// per-lane v16h B layout (f16). Tile t = kt*2 + nt; per lane 16 contiguous f16.
// B layout: lane L -> col N = (L&15); element e -> K = kt*32 + (e<8 ? 0 : 16) + 8*(L>>4) + (e&7).
// One wave (32 threads) per tile.
// ---------------------------------------------------------------------------
__global__ void k_pack_b(const float* __restrict__ W, _Float16* __restrict__ out,
                         int ntile2) {
    const int t = blockIdx.x;
    if (t >= ntile2) return;
    const int lane = threadIdx.x & 31;
    const int kt = t >> 1, nt = t & 1;
    const int n = lane & 15, g = lane >> 4;
    _Float16* dst = out + (size_t)t * 512 + lane * 16;
#pragma unroll
    for (int e = 0; e < 16; ++e) {
        const int k = kt * 32 + (e < 8 ? 0 : 16) + 8 * g + (e & 7);
        dst[e] = (_Float16)W[(size_t)k * HID + nt * 16 + n];
    }
}

// ---------------------------------------------------------------------------
// h = target_features @ W_i_a   [250K,8]@[8,32]
// ---------------------------------------------------------------------------
__global__ void k_init_h(const float* __restrict__ tf, const float* __restrict__ Wia,
                         float* __restrict__ h) {
    const int tid = blockIdx.x * blockDim.x + threadIdx.x;
    if (tid >= N_ATOMS * HID) return;
    const int a = tid >> 5;
    const int l = tid & 31;
    float acc = 0.f;
#pragma unroll
    for (int k = 0; k < FEAT; ++k)
        acc = fmaf(tf[(size_t)a * FEAT + k], Wia[k * HID + l], acc);
    h[tid] = acc;
}

// ---------------------------------------------------------------------------
// pre[a] = sum_j relu(edge_in[p_j] @ W_i_b),  p_j from b_scope (0 = pad row)
// One wave per atom; lane = hidden channel; edge projection recomputed on the
// fly (never materialize the 512MB msg tensor).
// ---------------------------------------------------------------------------
__global__ void k_edge_agg(const float* __restrict__ fdg, const float* __restrict__ rij,
                           const int* __restrict__ b_scope, const float* __restrict__ Wib,
                           float* __restrict__ pre) {
    const int wave = (blockIdx.x * blockDim.x + threadIdx.x) >> 5;
    const int lane = threadIdx.x & 31;
    if (wave >= N_ATOMS) return;  // uniform per-wave

    float wib[ALL_FDIM];
#pragma unroll
    for (int k = 0; k < ALL_FDIM; ++k) wib[k] = Wib[k * HID + lane];

    float acc = 0.f;
#pragma unroll
    for (int j = 0; j < MAX_NEI; ++j) {
        const int v = b_scope[(size_t)wave * MAX_NEI + j];  // wave-uniform
        if (v > 0) {
            const size_t p = (size_t)(v - 1);
            float x = 0.f;
            if (lane < FEAT)            x = fdg[p * FEAT + lane];
            else if (lane == FEAT)      x = rij[p];
            float y = 0.f;
#pragma unroll
            for (int k = 0; k < ALL_FDIM; ++k)
                y = fmaf(__shfl(x, k, 32), wib[k], y);
            acc += fmaxf(y, 0.f);
        }
    }
    pre[(size_t)wave * HID + lane] = acc;
}

// ---------------------------------------------------------------------------
// pre[a] = sum_j (h[su[p_j]] + h[sul[p_j]])  — gathers from L2-resident h.
// (em @ W_h then gather-sum == gather-sum then @ W_h by linearity.)
// ---------------------------------------------------------------------------
__global__ void k_pair_agg(const float* __restrict__ h, const int* __restrict__ b_scope,
                           const int* __restrict__ su, const int* __restrict__ sul,
                           float* __restrict__ pre) {
    const int wave = (blockIdx.x * blockDim.x + threadIdx.x) >> 5;
    const int lane = threadIdx.x & 31;
    if (wave >= N_ATOMS) return;

    float acc = 0.f;
#pragma unroll
    for (int j = 0; j < MAX_NEI; ++j) {
        const int v = b_scope[(size_t)wave * MAX_NEI + j];  // wave-uniform
        if (v > 0) {
            const size_t p  = (size_t)(v - 1);
            const int   i1 = su[p];    // wave-uniform -> scalar load
            const int   i2 = sul[p];
            acc += h[(size_t)i1 * HID + lane] + h[(size_t)i2 * HID + lane];
        }
    }
    pre[(size_t)wave * HID + lane] = acc;
}

// ---------------------------------------------------------------------------
// agg = pre @ W_h   via WMMA: 16-atom tiles, K=32 in one instruction.
// ---------------------------------------------------------------------------
__global__ void k_wh(const float* __restrict__ pre, const _Float16* __restrict__ pW,
                     float* __restrict__ out, int ntiles) {
    const int wave = threadIdx.x >> 5;
    const int lane = threadIdx.x & 31;
    int tile = blockIdx.x * (blockDim.x >> 5) + wave;
    const bool valid = tile < ntiles;
    if (!valid) tile = 0;
    const int row0 = tile << 4;

    const v16h a = load_A_rowmajor(pre, row0, lane);
#pragma unroll
    for (int nt = 0; nt < 2; ++nt) {
        const v16h b = load_B_packed(pW, nt, lane);
        v8f c = {};
        c = wmma_acc(a, b, c);
        store_D(out, row0, nt * 16, lane, c, valid);
    }
}

// ---------------------------------------------------------------------------
// Fused GRU cell over 16-atom tiles, WMMA for all three gate matmuls:
//   z = sig([agg,h]Wz+bz), r = sig([agg,h]Wr+br), hc = tanh([agg,r*h]Wh+bh)
//   h = (1-z)h + z*hc     (in place)
// r*h is transposed D-layout -> A-layout through per-wave LDS staging.
// Packed weight tile order per depth: tile = kt*2 + nt (kt: K chunk 0/1).
// ---------------------------------------------------------------------------
__global__ void k_gru(float* __restrict__ h, const float* __restrict__ agg,
                      const _Float16* __restrict__ pWz, const _Float16* __restrict__ pWr,
                      const _Float16* __restrict__ pWh, const float* __restrict__ bz,
                      const float* __restrict__ br, const float* __restrict__ bh,
                      int ntiles) {
    __shared__ float stage[8][16][36];  // [wave][row m][col n]; 36 keeps 16B align
    const int wave = threadIdx.x >> 5;
    const int lane = threadIdx.x & 31;
    int tile = blockIdx.x * (blockDim.x >> 5) + wave;
    const bool valid = tile < ntiles;
    if (!valid) tile = 0;
    const int row0 = tile << 4;
    const int n = lane & 15;
    const int g = lane >> 4;

    // A operands (lane = atom row, K across elements)
    const v16h a_agg = load_A_rowmajor(agg, row0, lane);
    const v16h a_h   = load_A_rowmajor(h,   row0, lane);

    float zv[2][8], hD[2][8];

#pragma unroll
    for (int nt = 0; nt < 2; ++nt) {
        const int col = nt * 16 + n;
        // z-gate: K chunk 0 = agg, K chunk 1 = h
        v8f cz = {};
        cz = wmma_acc(a_agg, load_B_packed(pWz, nt,     lane), cz);
        cz = wmma_acc(a_h,   load_B_packed(pWz, 2 + nt, lane), cz);
        // r-gate
        v8f crr = {};
        crr = wmma_acc(a_agg, load_B_packed(pWr, nt,     lane), crr);
        crr = wmma_acc(a_h,   load_B_packed(pWr, 2 + nt, lane), crr);

        const float bzv = bz[col];
        const float brv = br[col];
#pragma unroll
        for (int r = 0; r < 8; ++r) {
            const int m = r + 8 * g;
            const float hval = h[(size_t)(row0 + m) * HID + col];  // D-layout read
            hD[nt][r] = hval;
            const float z  = 1.f / (1.f + __expf(-(cz[r]  + bzv)));
            const float rr = 1.f / (1.f + __expf(-(crr[r] + brv)));
            zv[nt][r] = z;
            stage[wave][m][col] = rr * hval;  // r*h in [row][col] form
        }
    }
    __syncthreads();

    // Re-read r*h in A layout: per lane two contiguous 8-float runs -> b128 LDS
    v16h a_rh;
    {
        const int m = lane & 15;
        const float4* srow = (const float4*)(&stage[wave][m][0]);
        const float4 x0 = srow[2 * g];
        const float4 x1 = srow[2 * g + 1];
        const float4 x2 = srow[4 + 2 * g];
        const float4 x3 = srow[5 + 2 * g];
        a_rh[0]  = (_Float16)x0.x; a_rh[1]  = (_Float16)x0.y;
        a_rh[2]  = (_Float16)x0.z; a_rh[3]  = (_Float16)x0.w;
        a_rh[4]  = (_Float16)x1.x; a_rh[5]  = (_Float16)x1.y;
        a_rh[6]  = (_Float16)x1.z; a_rh[7]  = (_Float16)x1.w;
        a_rh[8]  = (_Float16)x2.x; a_rh[9]  = (_Float16)x2.y;
        a_rh[10] = (_Float16)x2.z; a_rh[11] = (_Float16)x2.w;
        a_rh[12] = (_Float16)x3.x; a_rh[13] = (_Float16)x3.y;
        a_rh[14] = (_Float16)x3.z; a_rh[15] = (_Float16)x3.w;
    }

#pragma unroll
    for (int nt = 0; nt < 2; ++nt) {
        const int col = nt * 16 + n;
        v8f c = {};
        c = wmma_acc(a_agg, load_B_packed(pWh, nt,     lane), c);
        c = wmma_acc(a_rh,  load_B_packed(pWh, 2 + nt, lane), c);
        const float bhv = bh[col];
        if (valid) {
#pragma unroll
            for (int r = 0; r < 8; ++r) {
                const int m = r + 8 * g;
                const float hc = tanhf(c[r] + bhv);
                const float z  = zv[nt][r];
                h[(size_t)(row0 + m) * HID + col] = (1.f - z) * hD[nt][r] + z * hc;
            }
        }
    }
}

// ---------------------------------------------------------------------------
// mol_vecs[m] = sum_j padded_h[l_scope[m,j]]   (index 0 = zero row)
// ---------------------------------------------------------------------------
__global__ void k_molvec(const float* __restrict__ h, const int* __restrict__ l_scope,
                         float* __restrict__ out) {
    const int wave = (blockIdx.x * blockDim.x + threadIdx.x) >> 5;
    const int lane = threadIdx.x & 31;
    if (wave >= N_MOL) return;
    float acc = 0.f;
    for (int j = 0; j < MAX_ATOMS_PER_MOL; ++j) {
        const int v = l_scope[(size_t)wave * MAX_ATOMS_PER_MOL + j];  // wave-uniform
        if (v > 0) acc += h[(size_t)(v - 1) * HID + lane];
    }
    out[(size_t)wave * HID + lane] = acc;
}

// ---------------------------------------------------------------------------
extern "C" void kernel_launch(void* const* d_in, const int* in_sizes, int n_in,
                              void* d_out, int out_size, void* d_ws, size_t ws_size,
                              hipStream_t stream) {
    const float* tf  = (const float*)d_in[0];
    const float* fdg = (const float*)d_in[1];
    const float* rij = (const float*)d_in[2];
    const int* b_scope = (const int*)d_in[3];
    const int* l_scope = (const int*)d_in[4];
    const int* su      = (const int*)d_in[5];
    const int* sul     = (const int*)d_in[6];
    const float* Wia = (const float*)d_in[7];
    const float* Wib = (const float*)d_in[8];
    const float* Whw = (const float*)d_in[9];
    const float* gWz = (const float*)d_in[10];
    const float* gWr = (const float*)d_in[11];
    const float* gWh = (const float*)d_in[12];
    const float* gbz = (const float*)d_in[13];
    const float* gbr = (const float*)d_in[14];
    const float* gbh = (const float*)d_in[15];

    float* h   = (float*)d_ws;                       // 32 MB
    float* pre = h   + (size_t)N_ATOMS * HID;        // 32 MB
    float* agg = pre + (size_t)N_ATOMS * HID;        // 32 MB
    // Packed f16 B tiles after the fp32 buffers:
    _Float16* pWh = (_Float16*)(agg + (size_t)N_ATOMS * HID);
    _Float16* pWz = pWh + 2 * 512;                   // [3 depths][4 tiles][512]
    _Float16* pWr = pWz + 12 * 512;
    _Float16* pWg = pWr + 12 * 512;

    const int NT = N_ATOMS / 16;                     // 15625 exact 16-row tiles
    const int tile_blocks = (NT + 7) / 8;            // 8 waves / 256-thread block
    const int atom_blocks = (N_ATOMS + 7) / 8;       // 1 wave per atom

    // Pack weights into WMMA B layout (f16). gru weights are [3,64,32]
    // contiguous == [192,32]: 12 tiles, depth-major, matching per-depth offsets.
    k_pack_b<<<2, 32, 0, stream>>>(Whw, pWh, 2);
    k_pack_b<<<12, 32, 0, stream>>>(gWz, pWz, 12);
    k_pack_b<<<12, 32, 0, stream>>>(gWr, pWr, 12);
    k_pack_b<<<12, 32, 0, stream>>>(gWh, pWg, 12);

    k_init_h<<<(N_ATOMS * HID + 255) / 256, 256, 0, stream>>>(tf, Wia, h);
    k_edge_agg<<<atom_blocks, 256, 0, stream>>>(fdg, rij, b_scope, Wib, pre);
    k_wh<<<tile_blocks, 256, 0, stream>>>(pre, pWh, agg, NT);

    for (int d = 0; d < DEPTH; ++d) {
        k_gru<<<tile_blocks, 256, 0, stream>>>(h, agg,
                                               pWz + (size_t)d * 2048,
                                               pWr + (size_t)d * 2048,
                                               pWg + (size_t)d * 2048,
                                               gbz + (size_t)d * HID,
                                               gbr + (size_t)d * HID,
                                               gbh + (size_t)d * HID, NT);
        if (d < DEPTH - 1) {  // final agg is unused by the reference
            k_pair_agg<<<atom_blocks, 256, 0, stream>>>(h, b_scope, su, sul, pre);
            k_wh<<<tile_blocks, 256, 0, stream>>>(pre, pWh, agg, NT);
        }
    }

    k_molvec<<<(N_MOL + 7) / 8, 256, 0, stream>>>(h, l_scope, (float*)d_out);
}
